// RNNWithMatrixParameter_54425825575661
// MI455X (gfx1250) — matrix operand order repaired and checked by reading, not compiled, not run on hardware
//
#include <hip/hip_runtime.h>
#include <hip/hip_bf16.h>

typedef __attribute__((ext_vector_type(16))) _Float16 v16h;
typedef __attribute__((ext_vector_type(8)))  float    v8f;

#define BB     32
#define EE     256
#define VV     32000
#define NSTEPC 128
#define MROWS  (BB * NSTEPC)   // 4096
#define NSLICE 10
#define SLICE  3200            // 10 * 3200 = 32000, 3200 % 64 == 0

// ---------------------------------------------------------------------------
// Kernel 1: transpose vocab_w (E x V, fp32) -> vocabT (V x E, f16), LDS-tiled
// ---------------------------------------------------------------------------
__global__ __launch_bounds__(256)
void transpose_vocab_kernel(const float* __restrict__ w, _Float16* __restrict__ wt) {
    __shared__ float tile[32][33];
    const int v0 = blockIdx.x * 32;
    const int k0 = blockIdx.y * 32;
    const int tx = threadIdx.x;      // 0..31 (v within tile on read)
    const int ty = threadIdx.y;      // 0..7
    #pragma unroll
    for (int j = 0; j < 32; j += 8)
        tile[ty + j][tx] = w[(size_t)(k0 + ty + j) * VV + (v0 + tx)];
    __syncthreads();
    #pragma unroll
    for (int j = 0; j < 32; j += 8)
        wt[(size_t)(v0 + ty + j) * EE + (k0 + tx)] = (_Float16)tile[tx][ty + j];
}

// ---------------------------------------------------------------------------
// Kernel 2: recurrence. One workgroup (256 threads) per batch element.
// trans (256x256 fp32 = 256KB) lives in LDS (CDNA5: 320KB per WG).
// ---------------------------------------------------------------------------
__global__ __launch_bounds__(256)
void recurrence_kernel(const float* __restrict__ latent, const int* __restrict__ zi,
                       float* __restrict__ zs_f32, _Float16* __restrict__ zs_f16) {
    extern __shared__ float smem[];
    float* trans = smem;                 // 65536 floats
    float* zn    = smem + EE * EE;       // 256 floats
    float* red   = zn + EE;              // 8 floats (sum)
    float* red2  = red + 8;              // 8 floats (sumsq)

    const int b    = blockIdx.x;
    const int f    = threadIdx.x;        // 0..255
    const int lane = f & 31;
    const int wid  = f >> 5;

    const float* src = latent + (size_t)zi[b] * (EE * EE);
    for (int i = f; i < EE * EE; i += 256) trans[i] = src[i];
    __syncthreads();

    float z = trans[f];                  // z0 = trans[0, f]

    for (int step = 0; step < NSTEPC; ++step) {
        // --- std(z, ddof=1) over the 256 elements ---
        float v = z, v2 = z * z;
        #pragma unroll
        for (int off = 16; off; off >>= 1) {
            v  += __shfl_xor(v,  off);
            v2 += __shfl_xor(v2, off);
        }
        if (lane == 0) { red[wid] = v; red2[wid] = v2; }
        __syncthreads();                 // B1
        float sum = 0.f, sq = 0.f;
        #pragma unroll
        for (int i = 0; i < 8; ++i) { sum += red[i]; sq += red2[i]; }
        const float mean = sum * (1.0f / EE);
        float var = (sq - sum * mean) * (1.0f / (EE - 1));
        var = var > 0.f ? var : 0.f;
        const float scale = 0.113f / (1e-5f + sqrtf(var));

        zn[f] = z * scale;
        __syncthreads();                 // B2

        // --- emit[f] = sum_e zn[e] * trans[e][f]  (lane-consecutive LDS reads) ---
        float acc = 0.f;
        #pragma unroll 8
        for (int e = 0; e < EE; ++e)
            acc = fmaf(zn[e], trans[e * EE + f], acc);

        z = acc;
        const size_t ro = ((size_t)b * NSTEPC + step) * EE + f;
        zs_f32[ro] = acc;
        zs_f16[ro] = (_Float16)acc;
        // red/zn rewrites next iteration are fenced by B1/B2 above.
    }
}

// ---------------------------------------------------------------------------
// Kernel 3: WMMA GEMM (32-row block x 3200-col slice) with fused online
// log-sum-exp epilogue. 8 waves: 2 (M) x 4 (N), each wave a 16x16 tile.
// ---------------------------------------------------------------------------
__global__ __launch_bounds__(256)
void gemm_lse_kernel(const _Float16* __restrict__ zsA, const _Float16* __restrict__ vocabT,
                     const float* __restrict__ vocab_b,
                     float* __restrict__ part_m, float* __restrict__ part_s) {
    __shared__ _Float16 As[32 * EE];     // 16 KB
    __shared__ float redm[32][4];
    __shared__ float reds[32][4];

    const int tid  = threadIdx.x;
    const int lane = tid & 31;
    const int wid  = tid >> 5;
    const int mi   = wid >> 2;           // 0..1  (M direction)
    const int ni   = wid & 3;            // 0..3  (N direction)
    const int rowBase   = blockIdx.x * 32;
    const int sliceBase = blockIdx.y * SLICE;

    // cooperative A-tile load (32 rows x 256 K, contiguous)
    {
        const uint4* srcv = (const uint4*)(zsA + (size_t)rowBase * EE);
        uint4* dstv = (uint4*)As;
        #pragma unroll
        for (int i = tid; i < (32 * EE) / 8; i += 256) dstv[i] = srcv[i];
    }
    __syncthreads();

    // build the 8 K-chunk A fragments (invariant over the N loop)
    const int half = lane >> 4;          // 0/1
    const int mrow = lane & 15;
    v16h afr[8];
    #pragma unroll
    for (int kc = 0; kc < 8; ++kc) {
        v16h a;
        #pragma unroll
        for (int p = 0; p < 8; ++p) {
            const int lk  = 2 * p + 16 * half;
            const int idx = (mi * 16 + mrow) * EE + kc * 32 + lk;
            a[2 * p]     = As[idx];
            a[2 * p + 1] = As[idx + 1];
        }
        afr[kc] = a;
    }

    // per-lane online softmax state: 8 rows per lane (C layout)
    float mm[8], ss[8];
    #pragma unroll
    for (int i = 0; i < 8; ++i) { mm[i] = -1e30f; ss[i] = 0.f; }

    const int colLane = lane & 15;
    for (int nc = sliceBase + ni * 16; nc < sliceBase + SLICE; nc += 64) {
        const int col = nc + colLane;
        const v16h* bp = (const v16h*)(vocabT + (size_t)col * EE + half * 16);
        __builtin_prefetch(vocabT + (size_t)(col + 64) * EE, 0, 1);

        v8f c = {};
        #pragma unroll
        for (int kc = 0; kc < 8; ++kc) {
            v16h bfr = bp[kc * 2];
            c = __builtin_amdgcn_wmma_f32_16x16x32_f16(
                    false, afr[kc], false, bfr, (short)0, c, false, false);
        }

        const float bias = vocab_b[col];
        #pragma unroll
        for (int i = 0; i < 8; ++i) {
            const float l  = c[i] + bias;
            const float nm = fmaxf(mm[i], l);
            ss[i] = ss[i] * __expf(mm[i] - nm) + __expf(l - nm);
            mm[i] = nm;
        }
    }

    // butterfly-merge (m,s) across the 16 lanes of each half
    #pragma unroll
    for (int i = 0; i < 8; ++i) {
        #pragma unroll
        for (int off = 1; off < 16; off <<= 1) {
            const float om = __shfl_xor(mm[i], off);
            const float os = __shfl_xor(ss[i], off);
            const float nm = fmaxf(mm[i], om);
            ss[i] = ss[i] * __expf(mm[i] - nm) + os * __expf(om - nm);
            mm[i] = nm;
        }
    }
    if ((lane & 15) == 0) {
        #pragma unroll
        for (int i = 0; i < 8; ++i) {
            const int rl = mi * 16 + half * 8 + i;
            redm[rl][ni] = mm[i];
            reds[rl][ni] = ss[i];
        }
    }
    __syncthreads();

    // merge the 4 N-direction waves, emit per-(row, slice) partial
    if (tid < 32) {
        float M = -1e30f, S = 0.f;
        #pragma unroll
        for (int j = 0; j < 4; ++j) {
            const float om = redm[tid][j];
            const float os = reds[tid][j];
            const float nm = fmaxf(M, om);
            S = S * __expf(M - nm) + os * __expf(om - nm);
            M = nm;
        }
        const size_t o = (size_t)(rowBase + tid) * NSLICE + blockIdx.y;
        part_m[o] = M;
        part_s[o] = S;
    }
}

// ---------------------------------------------------------------------------
// Kernel 4: merge slice partials, compute exact fp32 target logit, emit yp.
// One wave per output row.
// ---------------------------------------------------------------------------
__global__ __launch_bounds__(256)
void combine_kernel(const float* __restrict__ part_m, const float* __restrict__ part_s,
                    const float* __restrict__ zs_f32, const float* __restrict__ vocab_w,
                    const float* __restrict__ vocab_b, const int* __restrict__ y,
                    float* __restrict__ out) {
    const int lane = threadIdx.x & 31;
    const int wid  = threadIdx.x >> 5;
    const int r    = blockIdx.x * 8 + wid;        // 0..4095

    float M = -1e30f, S = 0.f;
    #pragma unroll
    for (int j = 0; j < NSLICE; ++j) {
        const float om = part_m[(size_t)r * NSLICE + j];
        const float os = part_s[(size_t)r * NSLICE + j];
        const float nm = fmaxf(M, om);
        S = S * __expf(M - nm) + os * __expf(om - nm);
        M = nm;
    }
    const float lse = M + __logf(S);

    const int col = y[r];
    float acc = 0.f;
    #pragma unroll
    for (int j = 0; j < 8; ++j) {
        const int k = lane + 32 * j;
        acc = fmaf(zs_f32[(size_t)r * EE + k], vocab_w[(size_t)k * VV + col], acc);
    }
    #pragma unroll
    for (int off = 16; off; off >>= 1) acc += __shfl_xor(acc, off);

    if (lane == 0) out[r] = acc + vocab_b[col] - lse;
}

// ---------------------------------------------------------------------------
extern "C" void kernel_launch(void* const* d_in, const int* in_sizes, int n_in,
                              void* d_out, int out_size, void* d_ws, size_t ws_size,
                              hipStream_t stream) {
    const float* latent  = (const float*)d_in[0];   // (T, E*E)
    const float* vocab_w = (const float*)d_in[1];   // (E, V)
    const float* vocab_b = (const float*)d_in[2];   // (V,)
    const int*   zi      = (const int*)d_in[3];     // (B,)
    const int*   y       = (const int*)d_in[4];     // (B, NSTEP)
    float*       out     = (float*)d_out;           // (B, NSTEP)

    char* ws = (char*)d_ws;
    size_t o = 0;
    _Float16* vocabT = (_Float16*)(ws + o); o += (size_t)VV * EE * sizeof(_Float16);
    o = (o + 255) & ~(size_t)255;
    float*    zs32   = (float*)(ws + o);    o += (size_t)MROWS * EE * sizeof(float);
    o = (o + 255) & ~(size_t)255;
    _Float16* zs16   = (_Float16*)(ws + o); o += (size_t)MROWS * EE * sizeof(_Float16);
    o = (o + 255) & ~(size_t)255;
    float*    pm     = (float*)(ws + o);    o += (size_t)MROWS * NSLICE * sizeof(float);
    float*    ps     = (float*)(ws + o);    o += (size_t)MROWS * NSLICE * sizeof(float);

    // 1) vocab_w fp32 (E,V) -> f16 transposed (V,E)
    transpose_vocab_kernel<<<dim3(VV / 32, EE / 32), dim3(32, 8), 0, stream>>>(vocab_w, vocabT);

    // 2) serial recurrence, trans matrix resident in 256KB LDS per WG
    const size_t recLds = (size_t)(EE * EE + EE + 16) * sizeof(float);
    recurrence_kernel<<<BB, 256, recLds, stream>>>(latent, zi, zs32, zs16);

    // 3) WMMA GEMM + fused online log-sum-exp partials
    gemm_lse_kernel<<<dim3(MROWS / 32, NSLICE), 256, 0, stream>>>(zs16, vocabT, vocab_b, pm, ps);

    // 4) merge partials + exact target logit -> yp
    combine_kernel<<<MROWS / 8, 256, 0, stream>>>(pm, ps, zs32, vocab_w, vocab_b, y, out);
}
